// MutualCrossAttention_23957327577279
// MI455X (gfx1250) — matrix operand order repaired and checked by reading, not compiled, not run on hardware
//
#include <hip/hip_runtime.h>

// Mutual cross-attention for MI455X (gfx1250, wave32, WMMA).
// out[b,i,:] = softmax(x1 x2^T/8)[i,:] @ x2  +  softmax(x2 x1^T/8)[i,:] @ x1
// B=4, N=4096, D=64, fp32 in/out; bf16 WMMA internally (f32 accum).
//
// v5: native f32->bf16 casts for P packing (hardware cvt instead of bit
//     twiddling); otherwise identical to v4: transposed softmax through LDS,
//     one attention pass per wave (pass A -> out, pass B -> partial + add),
//     async double-buffered block-shared K/V tiles, ones-WMMA row sums.

typedef __attribute__((ext_vector_type(16))) __bf16 v16bf;
typedef __attribute__((ext_vector_type(8)))  float  v8f;
typedef __attribute__((ext_vector_type(4)))  float  f32x4;
typedef __attribute__((ext_vector_type(4)))  unsigned int u32x4;
typedef int v4i_gcc __attribute__((vector_size(16)));  // matches builtin param

#define BB 4
#define NN 4096
#define DD 64

#ifndef __has_builtin
#define __has_builtin(x) 0
#endif

#if __has_builtin(__builtin_amdgcn_global_load_async_to_lds_b128)
#define USE_ASYNC_LDS 1
#else
#define USE_ASYNC_LDS 0
#endif

__device__ __forceinline__ v8f wmma_bf16(v16bf a, v16bf b, v8f c) {
  // (neg_a, A, neg_b, B, c_mod, C, reuse_a, reuse_b)
  return __builtin_amdgcn_wmma_f32_16x16x32_bf16(false, a, false, b, (short)0, c,
                                                 false, false);
}

// Build a 16-element bf16 fragment from two 16-byte chunks (global or LDS).
__device__ __forceinline__ v16bf ld_frag(const unsigned short* p0,
                                         const unsigned short* p1) {
  union { u32x4 u[2]; v16bf v; } t;
  t.u[0] = *reinterpret_cast<const u32x4*>(p0);
  t.u[1] = *reinterpret_cast<const u32x4*>(p1);
  return t.v;
}

__device__ __forceinline__ f32x4 vmax4(f32x4 a, f32x4 b) {
  f32x4 r;
  r.x = fmaxf(a.x, b.x); r.y = fmaxf(a.y, b.y);
  r.z = fmaxf(a.z, b.z); r.w = fmaxf(a.w, b.w);
  return r;
}

#if USE_ASYNC_LDS
// 16-byte async global->LDS copy. Address-space casts go through integers:
// flat LDS addr low 32 bits are exactly the AS(3) offset (aperture rules).
__device__ __forceinline__ void async_cp16(const void* g, void* l) {
  v4i_gcc __attribute__((address_space(1)))* gp =
      (v4i_gcc __attribute__((address_space(1)))*)(unsigned long long)g;
  v4i_gcc __attribute__((address_space(3)))* lp =
      (v4i_gcc __attribute__((address_space(3)))*)(unsigned int)(unsigned long long)l;
  __builtin_amdgcn_global_load_async_to_lds_b128(gp, lp, 0, 0);
}

__device__ __forceinline__ void wait_async0() {
#if __has_builtin(__builtin_amdgcn_s_wait_asynccnt)
  __builtin_amdgcn_s_wait_asynccnt(0);
#else
  asm volatile("s_wait_asynccnt 0x0" ::: "memory");
#endif
}
#endif  // USE_ASYNC_LDS

// ---------------------------------------------------------------------------
// Prologue: fp32 -> bf16, plus per-batch transposed copies [b][d][n].
// ---------------------------------------------------------------------------
__global__ __launch_bounds__(256) void mca_convert_bf16(
    const float* __restrict__ x1, const float* __restrict__ x2,
    __bf16* __restrict__ x1b, __bf16* __restrict__ x2b,
    __bf16* __restrict__ x1t, __bf16* __restrict__ x2t) {
  int idx = blockIdx.x * 256 + threadIdx.x;
  if (idx >= BB * NN * DD) return;
  int d = idx & (DD - 1);
  int n = (idx / DD) & (NN - 1);
  int b = idx / (NN * DD);
  __bf16 ab = (__bf16)x1[idx];
  __bf16 cb = (__bf16)x2[idx];
  x1b[idx] = ab;
  x2b[idx] = cb;
  int t = (b * DD + d) * NN + n;
  x1t[t] = ab;
  x2t[t] = cb;
}

// ---------------------------------------------------------------------------
// Flash attention, ONE pass per wave (pass A -> out, pass B -> partial).
// 4 waves (128 threads) per block; all waves share batch + pass, so K/V tiles
// are block-shared through double-buffered async LDS staging.
// ---------------------------------------------------------------------------
__global__ __launch_bounds__(128) void mca_flash_wmma(
    const unsigned short* __restrict__ x1b, const unsigned short* __restrict__ x2b,
    const unsigned short* __restrict__ x1t, const unsigned short* __restrict__ x2t,
    float* __restrict__ out, float* __restrict__ part) {
#if USE_ASYNC_LDS
  // Double-buffered block-shared tiles; rows padded to dodge bank conflicts.
  __shared__ unsigned short kbuf[2][32][72];  // [key row][d], 9.0 KB
  __shared__ unsigned short vbuf[2][64][40];  // V^T: [d][key], 10.0 KB
#endif
  // Per-wave f32 S bounce (C-layout write, A-layout read) + scale broadcast.
  __shared__ float sbuf[4][16][36];           // 9.0 KB
  __shared__ float scb[4][16];                // 256 B

  const int tid  = threadIdx.x;
  const int wave = tid >> 5;
  const int lane = tid & 31;
  const int r = lane & 15;   // column within C/D layout, row within A layout
  const int h = lane >> 4;   // lane half
  const int bi   = blockIdx.x;            // 0..511
  const int pass = bi & 1;
  const int tile = (bi >> 1) * 4 + wave;  // 0..1023
  const int b = tile >> 8;                // 256 tiles per batch
  const int rowbase = (tile & 255) << 4;  // 16 query rows per wave

  // All-ones B fragment: one WMMA computes the row-sums of P (replicated
  // across all 16 columns -> no cross-lane reduction needed for l).
  v16bf ones;
#pragma unroll
  for (int i = 0; i < 16; ++i) ones[i] = (__bf16)1.0f;

  const unsigned short* Q  = pass ? x2b : x1b;
  const unsigned short* K  = pass ? x1b : x2b;
  const unsigned short* VT = pass ? x1t : x2t;
  const unsigned short* Kb = K  + b * NN * DD;
  const unsigned short* Vb = VT + b * DD * NN;

  // Q A-fragments (16x32 bf16 each): lane holds row r; element e of frag f
  // maps to column 32f + 16h + e.
  const unsigned short* qrow = Q + (b * NN + rowbase + r) * DD;
  v16bf qf0 = ld_frag(qrow + 16 * h,      qrow + 16 * h + 8);
  v16bf qf1 = ld_frag(qrow + 32 + 16 * h, qrow + 40 + 16 * h);

  float m = -__builtin_inff();   // running row-max (row r, both halves agree)
  v8f lacc = {};
  v8f acc0 = {}, acc1 = {}, acc2 = {}, acc3 = {};

#if USE_ASYNC_LDS
  int buf = 0;
  {
    // Prime buffer 0 with the j=0 tiles: 512 16B chunks over 128 threads.
    int c0 = tid, c1 = tid + 128;
    async_cp16(Kb + (c0 >> 3) * DD + (c0 & 7) * 8, &kbuf[0][c0 >> 3][(c0 & 7) * 8]);
    async_cp16(Kb + (c1 >> 3) * DD + (c1 & 7) * 8, &kbuf[0][c1 >> 3][(c1 & 7) * 8]);
    async_cp16(Vb + (c0 >> 2) * NN + (c0 & 3) * 8, &vbuf[0][c0 >> 2][(c0 & 3) * 8]);
    async_cp16(Vb + (c1 >> 2) * NN + (c1 & 3) * 8, &vbuf[0][c1 >> 2][(c1 & 3) * 8]);
    wait_async0();
    __syncthreads();
  }
#endif

  for (int j = 0; j < NN; j += 32) {
#if USE_ASYNC_LDS
    if (j + 32 < NN) {  // prefetch next tile into the other buffer
      int nb = buf ^ 1;
      int jn = j + 32;
      int c0 = tid, c1 = tid + 128;
      const unsigned short* kn = Kb + jn * DD;
      async_cp16(kn + (c0 >> 3) * DD + (c0 & 7) * 8, &kbuf[nb][c0 >> 3][(c0 & 7) * 8]);
      async_cp16(kn + (c1 >> 3) * DD + (c1 & 7) * 8, &kbuf[nb][c1 >> 3][(c1 & 7) * 8]);
      async_cp16(Vb + (c0 >> 2) * NN + jn + (c0 & 3) * 8, &vbuf[nb][c0 >> 2][(c0 & 3) * 8]);
      async_cp16(Vb + (c1 >> 2) * NN + jn + (c1 & 3) * 8, &vbuf[nb][c1 >> 2][(c1 & 3) * 8]);
    }
    const unsigned short* k0 = &kbuf[buf][r][16 * h];
    const unsigned short* k1 = &kbuf[buf][16 + r][16 * h];
    const unsigned short* v0 = &vbuf[buf][0  + r][16 * h];
    const unsigned short* v1 = &vbuf[buf][16 + r][16 * h];
    const unsigned short* v2 = &vbuf[buf][32 + r][16 * h];
    const unsigned short* v3 = &vbuf[buf][48 + r][16 * h];
#else
    const unsigned short* k0 = Kb + (j + r) * DD + 16 * h;
    const unsigned short* k1 = Kb + (j + 16 + r) * DD + 16 * h;
    const unsigned short* v0 = Vb + (0  + r) * NN + j + 16 * h;
    const unsigned short* v1 = Vb + (16 + r) * NN + j + 16 * h;
    const unsigned short* v2 = Vb + (32 + r) * NN + j + 16 * h;
    const unsigned short* v3 = Vb + (48 + r) * NN + j + 16 * h;
    if (j + 32 < NN) {
      __builtin_prefetch(Kb + (j + 32 + r) * DD, 0, 0);
      __builtin_prefetch(Vb + r * NN + j + 32, 0, 0);
    }
#endif
    // Load every fragment for this step up front.
    v16bf kf00 = ld_frag(k0,      k0 + 8);
    v16bf kf01 = ld_frag(k0 + 32, k0 + 40);
    v16bf kf10 = ld_frag(k1,      k1 + 8);
    v16bf kf11 = ld_frag(k1 + 32, k1 + 40);
    v16bf vf0  = ld_frag(v0, v0 + 8);
    v16bf vf1  = ld_frag(v1, v1 + 8);
    v16bf vf2  = ld_frag(v2, v2 + 8);
    v16bf vf3  = ld_frag(v3, v3 + 8);

    v8f z = {};
    v8f S0 = wmma_bf16(qf0, kf00, z);
    S0     = wmma_bf16(qf1, kf01, S0);
    v8f S1 = wmma_bf16(qf0, kf10, z);
    S1     = wmma_bf16(qf1, kf11, S1);

    // --- transpose raw S through LDS: C-layout write, A-layout read ---
#pragma unroll
    for (int v = 0; v < 8; ++v) {
      sbuf[wave][v + 8 * h][r]      = S0[v];
      sbuf[wave][v + 8 * h][16 + r] = S1[v];
    }
    asm volatile("s_wait_dscnt 0x0" ::: "memory");

    const float* srow = &sbuf[wave][r][0];
    f32x4 a0 = *reinterpret_cast<const f32x4*>(srow + 16 * h);
    f32x4 a1 = *reinterpret_cast<const f32x4*>(srow + 16 * h + 4);
    f32x4 a2 = *reinterpret_cast<const f32x4*>(srow + 16 * h + 8);
    f32x4 a3 = *reinterpret_cast<const f32x4*>(srow + 16 * h + 12);

    // Row max: in-register tree over 16 values + one cross-half shuffle.
    f32x4 mx = vmax4(vmax4(a0, a1), vmax4(a2, a3));
    float t = fmaxf(fmaxf(mx.x, mx.y), fmaxf(mx.z, mx.w));
    t = fmaxf(t, __shfl_xor(t, 16));
    float mn = fmaxf(m, t);
    float sc = __expf((m - mn) * 0.125f);  // 1/sqrt(64) folded into exp domain
    m = mn;
    scb[wave][r] = sc;                     // broadcast back to C-layout lanes
    float cmn = mn * 0.125f;

    // P = exp(0.125*s - 0.125*mn), packed straight into the A-layout fragment
    // with native f32->bf16 conversions.
    v16bf pf;
#pragma unroll
    for (int e = 0; e < 4; ++e) {
      pf[e]      = (__bf16)__expf(__builtin_fmaf(a0[e], 0.125f, -cmn));  // 0..3
      pf[4 + e]  = (__bf16)__expf(__builtin_fmaf(a1[e], 0.125f, -cmn));  // 4..7
      pf[8 + e]  = (__bf16)__expf(__builtin_fmaf(a2[e], 0.125f, -cmn));  // 8..11
      pf[12 + e] = (__bf16)__expf(__builtin_fmaf(a3[e], 0.125f, -cmn));  // 12..15
    }

    asm volatile("s_wait_dscnt 0x0" ::: "memory");
    f32x4 sA = *reinterpret_cast<const f32x4*>(&scb[wave][8 * h]);
    f32x4 sB = *reinterpret_cast<const f32x4*>(&scb[wave][8 * h + 4]);

    // Rescale accumulators by the per-row factor (C-layout rows v+8h).
#pragma unroll
    for (int v = 0; v < 4; ++v) {
      lacc[v] *= sA[v];     lacc[4 + v] *= sB[v];
      acc0[v] *= sA[v];     acc0[4 + v] *= sB[v];
      acc1[v] *= sA[v];     acc1[4 + v] *= sB[v];
      acc2[v] *= sA[v];     acc2[4 + v] *= sB[v];
      acc3[v] *= sA[v];     acc3[4 + v] *= sB[v];
    }

    lacc = wmma_bf16(pf, ones, lacc);  // row-sums of P, no lane reduction
    acc0 = wmma_bf16(pf, vf0, acc0);
    acc1 = wmma_bf16(pf, vf1, acc1);
    acc2 = wmma_bf16(pf, vf2, acc2);
    acc3 = wmma_bf16(pf, vf3, acc3);

#if USE_ASYNC_LDS
    wait_async0();     // next tile resident ...
    __syncthreads();   // ... in every wave; current tile reads all done
    buf ^= 1;
#endif
  }

  float* dst = (pass ? part : out) + (b * NN + rowbase) * DD;
#pragma unroll
  for (int v = 0; v < 8; ++v) {
    float inv = 1.0f / lacc[v];
    int row = v + 8 * h;
    dst[row * DD +      r] = acc0[v] * inv;
    dst[row * DD + 16 + r] = acc1[v] * inv;
    dst[row * DD + 32 + r] = acc2[v] * inv;
    dst[row * DD + 48 + r] = acc3[v] * inv;
  }
}

// ---------------------------------------------------------------------------
// Epilogue: out += partial (pass B contribution).
// ---------------------------------------------------------------------------
__global__ __launch_bounds__(256) void mca_add(const float* __restrict__ part,
                                               float* __restrict__ out) {
  int idx = blockIdx.x * 256 + threadIdx.x;
  if (idx < BB * NN * DD) out[idx] += part[idx];
}

// ---------------------------------------------------------------------------
extern "C" void kernel_launch(void* const* d_in, const int* in_sizes, int n_in,
                              void* d_out, int out_size, void* d_ws, size_t ws_size,
                              hipStream_t stream) {
  const float* x1 = (const float*)d_in[0];
  const float* x2 = (const float*)d_in[1];
  float* out = (float*)d_out;

  const int elems = BB * NN * DD;  // 1,048,576
  unsigned short* w   = (unsigned short*)d_ws;
  unsigned short* x1b = w;
  unsigned short* x2b = w + (size_t)elems;
  unsigned short* x1t = w + (size_t)2 * elems;
  unsigned short* x2t = w + (size_t)3 * elems;       // 8 MB bf16 scratch
  float* part = (float*)(w + (size_t)4 * elems);     // +4 MB f32 partial

  mca_convert_bf16<<<(elems + 255) / 256, 256, 0, stream>>>(
      x1, x2, (__bf16*)x1b, (__bf16*)x2b, (__bf16*)x1t, (__bf16*)x2t);
  // One pass per wave: 2 passes * B * (N/16) waves / 4 per block = 512 blocks.
  mca_flash_wmma<<<2 * (BB * (NN / 16)) / 4, 128, 0, stream>>>(x1b, x2b, x1t,
                                                               x2t, out, part);
  mca_add<<<(elems + 255) / 256, 256, 0, stream>>>(part, out);
}
